// customGNN_5531917877736
// MI455X (gfx1250) — matrix operand order repaired and checked by reading, not compiled, not run on hardware
//
#include <hip/hip_runtime.h>
#include <stdint.h>

typedef __attribute__((ext_vector_type(16))) _Float16 v16h;
typedef __attribute__((ext_vector_type(8)))  _Float16 v8h;
typedef __attribute__((ext_vector_type(8)))  float    v8f;

#define NEG_SLOPE 0.2f
#define ENC_NEG_INF 0x007FFFFFu  // enc(-inf)

// ---- helpers -------------------------------------------------------------

__device__ __forceinline__ unsigned encf(float f) {
    unsigned b = __float_as_uint(f);
    return (b & 0x80000000u) ? ~b : (b | 0x80000000u);
}
__device__ __forceinline__ float decf(unsigned k) {
    return __uint_as_float((k & 0x80000000u) ? (k & 0x7FFFFFFFu) : ~k);
}
__device__ __forceinline__ void edge_sd(const long long* ei, int e, int E,
                                        int& s, int& d) {
    if (e < E) { s = (int)ei[e]; d = (int)ei[(size_t)E + e]; }
    else       { s = d = e - E; }   // self loop appended after E edges
}

// ---- f32 -> f16 conversion ----------------------------------------------

__global__ void k_f32_to_f16(const float* __restrict__ src,
                             _Float16* __restrict__ dst, int n) {
    int i = blockIdx.x * blockDim.x + threadIdx.x;
    if (i < n) dst[i] = (_Float16)src[i];
}

// Pack [Wl | Wr] (each [K, HC] f32 row-major) TRANSPOSED into f16 Wt[NC2, K]:
// Wt[j*K + k] so a WMMA B-fragment (16 consecutive K at fixed col) is
// contiguous in memory.
__global__ void k_pack_weights(const float* __restrict__ Wl,
                               const float* __restrict__ Wr,
                               _Float16* __restrict__ Wt, int K, int HC) {
    int i = blockIdx.x * blockDim.x + threadIdx.x;
    if (i >= K * 2 * HC) return;
    int j = i / K, k = i % K;
    float v = (j < HC) ? Wl[(size_t)k * HC + j] : Wr[(size_t)k * HC + (j - HC)];
    Wt[i] = (_Float16)v;
}

// ---- WMMA GEMM: C[M, NC] = A[M,K] * B[K,NC] + bias(col) ------------------
// Bt is the transposed weight pack [NC, K]. One wave -> 16 rows x 64 cols
// (4 wmma accumulators); 8 waves/block = 128 rows. The block's 64x32 B panel
// is staged into LDS via async global->LDS loads (ASYNCcnt), double-buffered.

__global__ void k_gemm_xlr(const _Float16* __restrict__ A,
                           const _Float16* __restrict__ Bt,
                           const float* __restrict__ bl,
                           const float* __restrict__ br,
                           float* __restrict__ C,
                           int M, int K, int NC, int HC) {
    __shared__ _Float16 Bs[2][64 * 32];   // [col][k] within panel

    const int tid  = threadIdx.x;
    const int wave = tid >> 5;
    const int lane = tid & 31;
    const int row0 = blockIdx.x * 128 + wave * 16;
    const int col0 = blockIdx.y * 64;

    const int r    = lane & 15;        // A row within tile / B,C column
    const int half = lane >> 4;        // lane half selects K sub-range

    // staging: each of 256 threads moves 16B (8 halves): col = tid/4, k = tid%4*8
    const int ccol = tid >> 2;
    const int ck   = (tid & 3) * 8;
    const _Float16* gB = Bt + (size_t)(col0 + ccol) * K + ck;

    auto stage = [&](int buf, int kt) {
        const _Float16* src = gB + kt;
        unsigned ldsoff = (unsigned)(uintptr_t)&Bs[buf][ccol * 32 + ck];
        asm volatile("global_load_async_to_lds_b128 %0, %1, off"
                     :: "v"(ldsoff), "v"(src) : "memory");
    };

    stage(0, 0);

    v8f acc0 = {}, acc1 = {}, acc2 = {}, acc3 = {};
    const _Float16* arow = A + (size_t)(row0 + r) * K + half * 16;

    int buf = 0;
    for (int kt = 0; kt < K; kt += 32) {
        asm volatile("s_wait_asynccnt 0x0" ::: "memory");
        __syncthreads();                       // panel ready; prev readers done
        if (kt + 32 < K) stage(buf ^ 1, kt + 32);

        if (row0 < M) {
            // A fragment (16x32 f16, contiguous): lanes 0-15 K=0..15
            //                                     lanes16-31 K=16..31
            const _Float16* ap = arow + kt;
            __builtin_prefetch(ap + 64, 0, 3);          // WGP-scope prefetch
            v8h alo = *(const v8h*)(ap);
            v8h ahi = *(const v8h*)(ap + 8);
            v16h af;
#pragma unroll
            for (int i = 0; i < 8; ++i) { af[i] = alo[i]; af[i + 8] = ahi[i]; }

            // Load ALL four B fragments first so the DS loads pipeline and
            // the four WMMAs can issue back-to-back.
            const _Float16* bsb = &Bs[buf][(size_t)half * 16 + r * 32];
            v16h bf0 = *(const v16h*)(bsb + 0 * 16 * 32);
            v16h bf1 = *(const v16h*)(bsb + 1 * 16 * 32);
            v16h bf2 = *(const v16h*)(bsb + 2 * 16 * 32);
            v16h bf3 = *(const v16h*)(bsb + 3 * 16 * 32);

            acc0 = __builtin_amdgcn_wmma_f32_16x16x32_f16(
                false, af, false, bf0, (short)0, acc0, false, false);
            acc1 = __builtin_amdgcn_wmma_f32_16x16x32_f16(
                false, af, false, bf1, (short)0, acc1, false, false);
            acc2 = __builtin_amdgcn_wmma_f32_16x16x32_f16(
                false, af, false, bf2, (short)0, acc2, false, false);
            acc3 = __builtin_amdgcn_wmma_f32_16x16x32_f16(
                false, af, false, bf3, (short)0, acc3, false, false);
        }
        buf ^= 1;
    }

    if (row0 >= M) return;

    // Epilogue: C/D layout — VGPR i: M = i + 8*half; N = lane&15 (+16*t)
#pragma unroll
    for (int t = 0; t < 4; ++t) {
        const v8f* accp = t == 0 ? &acc0 : t == 1 ? &acc1 : t == 2 ? &acc2 : &acc3;
        int col = col0 + t * 16 + r;
        float bias = (col < HC) ? bl[col] : br[col - HC];
#pragma unroll
        for (int i = 0; i < 8; ++i) {
            int row = row0 + i + 8 * half;
            C[(size_t)row * NC + col] = (*accp)[i] + bias;
        }
    }
}

// ---- per-layer state init ------------------------------------------------

__global__ void k_init(unsigned* __restrict__ menc, float* __restrict__ den,
                       float* __restrict__ outb, int nh, int nout) {
    int i = blockIdx.x * blockDim.x + threadIdx.x;
    if (i < nh) { menc[i] = ENC_NEG_INF; den[i] = 0.0f; }
    if (i < nout) outb[i] = 0.0f;
}

// ---- edge score + segment max (wave per edge, float4 per lane) -----------

__global__ void k_score(const float* __restrict__ XLR,
                        const long long* __restrict__ ei,
                        const float* __restrict__ att,
                        float* __restrict__ sc, unsigned* __restrict__ menc,
                        int E, int ET, int H, int Cc, int NC2, int HC) {
    int e = blockIdx.x * 8 + (threadIdx.x >> 5);
    if (e >= ET) return;
    int lane = threadIdx.x & 31;
    int s, d;
    edge_sd(ei, e, E, s, d);
    const float* xls = XLR + (size_t)s * NC2;            // xl[src]
    const float* xrd = XLR + (size_t)d * NC2 + HC;       // xr[dst]
    for (int h = 0; h < H; ++h) {
        const float* xl = xls + h * Cc;
        const float* xr = xrd + h * Cc;
        const float* at = att + h * Cc;
        float p = 0.0f;
        for (int c = lane * 4; c < Cc; c += 128) {       // float4 per lane
            float4 a = *(const float4*)(xl + c);
            float4 b = *(const float4*)(xr + c);
            float4 w = *(const float4*)(at + c);
            float v0 = a.x + b.x, v1 = a.y + b.y, v2 = a.z + b.z, v3 = a.w + b.w;
            v0 = v0 > 0.0f ? v0 : NEG_SLOPE * v0;
            v1 = v1 > 0.0f ? v1 : NEG_SLOPE * v1;
            v2 = v2 > 0.0f ? v2 : NEG_SLOPE * v2;
            v3 = v3 > 0.0f ? v3 : NEG_SLOPE * v3;
            p += w.x * v0 + w.y * v1 + w.z * v2 + w.w * v3;
        }
#pragma unroll
        for (int o = 16; o > 0; o >>= 1) p += __shfl_xor(p, o, 32);
        if (lane == 0) {
            sc[(size_t)e * H + h] = p;
            atomicMax(&menc[(size_t)d * H + h], encf(p));
        }
    }
}

// ---- exp + segment sum (thread per edge*head) ----------------------------

__global__ void k_exp(float* __restrict__ sc, const unsigned* __restrict__ menc,
                      float* __restrict__ den, const long long* __restrict__ ei,
                      int E, int ET, int H) {
    int i = blockIdx.x * blockDim.x + threadIdx.x;
    if (i >= ET * H) return;
    int e = i / H, h = i % H;
    int s, d;
    edge_sd(ei, e, E, s, d);
    float m = decf(menc[(size_t)d * H + h]);
    float x = __expf(sc[i] - m);
    sc[i] = x;
    atomicAdd(&den[(size_t)d * H + h], x);
}

// ---- weighted aggregation (wave per edge, float4 per lane) ---------------

__global__ void k_aggregate(const float* __restrict__ XLR,
                            const long long* __restrict__ ei,
                            const float* __restrict__ sc,
                            const float* __restrict__ den,
                            float* __restrict__ outb,
                            int E, int ET, int H, int Cc, int NC2, int HC) {
    int e = blockIdx.x * 8 + (threadIdx.x >> 5);
    if (e >= ET) return;
    int lane = threadIdx.x & 31;
    int s, d;
    edge_sd(ei, e, E, s, d);
    for (int h = 0; h < H; ++h) {
        float alpha = sc[(size_t)e * H + h] / den[(size_t)d * H + h];
        const float* xl = XLR + (size_t)s * NC2 + h * Cc;
        float* op = outb + (size_t)d * HC + h * Cc;
        for (int c = lane * 4; c < Cc; c += 128) {       // float4 per lane
            float4 a = *(const float4*)(xl + c);
            atomicAdd(&op[c + 0], alpha * a.x);
            atomicAdd(&op[c + 1], alpha * a.y);
            atomicAdd(&op[c + 2], alpha * a.z);
            atomicAdd(&op[c + 3], alpha * a.w);
        }
    }
}

// ---- bias + ELU + downconvert for next layer -----------------------------

__global__ void k_finalize(const float* __restrict__ outb,
                           const float* __restrict__ bias,
                           _Float16* __restrict__ Anext,
                           float* __restrict__ outf,
                           int total, int HC, int last) {
    int i = blockIdx.x * blockDim.x + threadIdx.x;
    if (i >= total) return;
    int j = i % HC;
    float v = outb[i] + bias[j];
    if (last) {
        outf[i] = v;
    } else {
        v = v > 0.0f ? v : __expf(v) - 1.0f;             // elu
        Anext[i] = (_Float16)v;
    }
}

// ---- host orchestration --------------------------------------------------

extern "C" void kernel_launch(void* const* d_in, const int* in_sizes, int n_in,
                              void* d_out, int out_size, void* d_ws, size_t ws_size,
                              hipStream_t stream) {
    const float*     x  = (const float*)d_in[0];
    const long long* ei = (const long long*)d_in[1];

    const int Din = 256;
    const int N   = in_sizes[0] / Din;       // 10000
    const int E   = in_sizes[1] / 2;         // 160000
    const int ET  = E + N;                   // + self loops

    // workspace carve-out (all 256B aligned)
    char* ws = (char*)d_ws;
    size_t off = 0;
    auto carve = [&](size_t bytes) {
        char* p = ws + off;
        off = (off + bytes + 255) & ~(size_t)255;
        return p;
    };
    _Float16* A16  = (_Float16*)carve((size_t)N * 512 * sizeof(_Float16));
    _Float16* W16  = (_Float16*)carve((size_t)512 * 1024 * sizeof(_Float16));
    float*    XLR  = (float*)   carve((size_t)N * 1024 * sizeof(float));
    float*    OUTB = (float*)   carve((size_t)N * 512 * sizeof(float));
    float*    SC   = (float*)   carve((size_t)ET * 4 * sizeof(float));
    unsigned* MENC = (unsigned*)carve((size_t)N * 4 * sizeof(unsigned));
    float*    DEN  = (float*)   carve((size_t)N * 4 * sizeof(float));
    (void)ws_size; (void)n_in; (void)out_size;

    struct Layer { int K, H, C; };
    const Layer L[4] = {{256, 4, 128}, {512, 4, 128}, {512, 4, 128}, {512, 1, 256}};

    // layer-0 activation conversion: x -> f16
    {
        int n = N * Din;
        k_f32_to_f16<<<(n + 255) / 256, 256, 0, stream>>>(x, A16, n);
    }

    for (int i = 0; i < 4; ++i) {
        const int K = L[i].K, H = L[i].H, Cc = L[i].C;
        const int HC = H * Cc, NC2 = 2 * HC;
        const float* Wl   = (const float*)d_in[2 + 6 * i + 0];
        const float* bl   = (const float*)d_in[2 + 6 * i + 1];
        const float* Wr   = (const float*)d_in[2 + 6 * i + 2];
        const float* br   = (const float*)d_in[2 + 6 * i + 3];
        const float* att  = (const float*)d_in[2 + 6 * i + 4];
        const float* bias = (const float*)d_in[2 + 6 * i + 5];

        // pack [Wl|Wr] -> f16, transposed to [NC2, K]
        {
            int n = K * NC2;
            k_pack_weights<<<(n + 255) / 256, 256, 0, stream>>>(Wl, Wr, W16, K, HC);
        }
        // xl|xr = A @ [Wl|Wr] + [bl|br]   (WMMA f16 -> f32)
        {
            dim3 grid((N + 127) / 128, NC2 / 64);
            k_gemm_xlr<<<grid, 256, 0, stream>>>(A16, W16, bl, br, XLR, N, K, NC2, HC);
        }
        // per-layer reset of max / denom / accumulator
        {
            int n = N * HC;
            k_init<<<(n + 255) / 256, 256, 0, stream>>>(MENC, DEN, OUTB, N * H, n);
        }
        // edge phase
        k_score<<<(ET + 7) / 8, 256, 0, stream>>>(XLR, ei, att, SC, MENC,
                                                  E, ET, H, Cc, NC2, HC);
        k_exp<<<(ET * H + 255) / 256, 256, 0, stream>>>(SC, MENC, DEN, ei, E, ET, H);
        k_aggregate<<<(ET + 7) / 8, 256, 0, stream>>>(XLR, ei, SC, DEN, OUTB,
                                                      E, ET, H, Cc, NC2, HC);
        // bias (+ELU, + f16 downconvert for next layer) / final output
        {
            int n = N * HC;
            k_finalize<<<(n + 255) / 256, 256, 0, stream>>>(
                OUTB, bias, A16, (float*)d_out, n, HC, i == 3 ? 1 : 0);
        }
    }
}
